// DkNNModel_2894807958249
// MI455X (gfx1250) — matrix operand order repaired and checked by reading, not compiled, not run on hardware
//
#include <hip/hip_runtime.h>
#include <hip/hip_bf16.h>

// ---- problem constants (from reference) ----
#define DD       512
#define NTRAIN   100000
#define BQ       256
#define LL       2
#define KNN      75
#define NC       10
#define NBCALI   750

// ---- tiling ----
#define MBLK     32          // query rows per workgroup (2 x 16-row tiles)
#define NBLK     192         // train cols per workgroup (12 x 16-col tiles)
#define NPAD     100032      // ceil(100000/192)*192 = 521*192
#define AQ_STRIDE 520        // halfs: 1040B row stride -> b128 reads hit all 64 banks
#define BT_STRIDE 40         // halfs: 80B row stride -> conflict-free 16-lane b128 reads
#define TIE_CAP  2048

typedef __attribute__((ext_vector_type(16))) _Float16 v16h;
typedef __attribute__((ext_vector_type(8)))  _Float16 v8h;
typedef __attribute__((ext_vector_type(8)))  float    v8f;

// monotone map float -> sortable u16 (ascending). smaller key == nearer neighbor.
__device__ __forceinline__ unsigned short sortkey16(float x) {
    unsigned u = __float_as_uint(x);
    u ^= ((unsigned)(((int)u) >> 31)) | 0x80000000u;
    return (unsigned short)(u >> 16);
}

// ---- pass 1a: per-train-row inverse norms (wave per row) ----
__global__ __launch_bounds__(256) void prep_train(const float* __restrict__ tf,
                                                  float* __restrict__ invn) {
    int row  = blockIdx.x * 8 + (threadIdx.x >> 5);   // 0 .. L*NTRAIN-1 (exact)
    int lane = threadIdx.x & 31;
    const float* p = tf + (size_t)row * DD;
    float s = 0.f;
#pragma unroll
    for (int j = 0; j < DD / 32; ++j) { float v = p[lane + 32 * j]; s += v * v; }
#pragma unroll
    for (int m = 16; m; m >>= 1) s += __shfl_xor(s, m, 32);
    if (lane == 0) invn[row] = rsqrtf(s);
}

// ---- pass 1b: normalized queries -> f16 (wave per row) ----
__global__ __launch_bounds__(256) void prep_query(const float* __restrict__ qf,
                                                  _Float16* __restrict__ qh) {
    int row  = blockIdx.x * 8 + (threadIdx.x >> 5);   // 0..511 (exact)
    int lane = threadIdx.x & 31;
    const float* p = qf + (size_t)row * DD;
    float v[16];
    float s = 0.f;
#pragma unroll
    for (int j = 0; j < DD / 32; ++j) { v[j] = p[lane + 32 * j]; s += v[j] * v[j]; }
#pragma unroll
    for (int m = 16; m; m >>= 1) s += __shfl_xor(s, m, 32);
    float inv = rsqrtf(s);
#pragma unroll
    for (int j = 0; j < DD / 32; ++j)
        qh[(size_t)row * DD + lane + 32 * j] = (_Float16)(v[j] * inv);
}

__global__ __launch_bounds__(256) void init_counts(int* __restrict__ counts) {
    int i = blockIdx.x * 256 + threadIdx.x;
    if (i < BQ * NC) counts[i] = 0;
}

// ---- pass 2: WMMA GEMM  keys[l,b,n] = sortkey16(-q_hat . t_hat) ----
__global__ __launch_bounds__(256) void gemm_keys(const float* __restrict__ train,
                                                 const _Float16* __restrict__ qh,
                                                 const float* __restrict__ invn,
                                                 unsigned short* __restrict__ keys) {
    __shared__ __align__(16) _Float16 Aq[MBLK * AQ_STRIDE];
    __shared__ __align__(16) _Float16 Bt[2][NBLK * BT_STRIDE];

    const int l   = blockIdx.z;
    const int m0  = blockIdx.y * MBLK;
    const int n0  = blockIdx.x * NBLK;
    const int tid = threadIdx.x;

    // stage all 32 query rows (full K depth) into LDS, 32-bit copies
    {
        const unsigned* src = (const unsigned*)(qh + ((size_t)(l * BQ + m0)) * DD);
        for (int i = tid; i < MBLK * (DD / 2); i += 256) {
            int r = i >> 8, d2 = i & 255;                    // DD/2 == 256
            *(unsigned*)&Aq[r * AQ_STRIDE + d2 * 2] = src[r * (DD / 2) + d2];
        }
    }

    const int lane = tid & 31;
    const int w    = tid >> 5;
    const int mt   = w & 1;              // 0..1 : 16-row tile within MBLK
    const int ntb  = (w >> 1) * 3;       // 0,3,6,9 : this wave's three 16-col tiles
    const int hi   = lane >> 4;          // 0/1 half-wave
    const int ln   = lane & 15;

    // B-tile loader setup (divergence-free: clamp address, zero the scale instead)
    const int   ld_n0  = tid >> 1;           // chunk 0: rows 0..127
    const int   ld_n1  = 128 + (tid >> 1);   // chunk 1: rows 128..191 (tid < 128 only)
    const int   ld_kq  = (tid & 1) * 16;

    v8f acc0 = {}; v8f acc1 = {}; v8f acc2 = {};

    // preload K-chunk 0 into buffer 0
    {
#pragma unroll
        for (int c = 0; c < 2; ++c) {
            int n = c ? ld_n1 : ld_n0;
            if (c && tid >= 128) break;
            int gn  = n0 + n;
            int gnc = gn < NTRAIN ? gn : NTRAIN - 1;
            float inv = gn < NTRAIN ? invn[l * NTRAIN + gnc] : 0.f;
            const float* gp = train + (((size_t)l * NTRAIN + (size_t)gnc) * DD + ld_kq);
            float4 f0 = ((const float4*)gp)[0], f1 = ((const float4*)gp)[1];
            float4 f2 = ((const float4*)gp)[2], f3 = ((const float4*)gp)[3];
            _Float16* dst = &Bt[0][n * BT_STRIDE + ld_kq];
            dst[0]  = (_Float16)(f0.x * inv); dst[1]  = (_Float16)(f0.y * inv);
            dst[2]  = (_Float16)(f0.z * inv); dst[3]  = (_Float16)(f0.w * inv);
            dst[4]  = (_Float16)(f1.x * inv); dst[5]  = (_Float16)(f1.y * inv);
            dst[6]  = (_Float16)(f1.z * inv); dst[7]  = (_Float16)(f1.w * inv);
            dst[8]  = (_Float16)(f2.x * inv); dst[9]  = (_Float16)(f2.y * inv);
            dst[10] = (_Float16)(f2.z * inv); dst[11] = (_Float16)(f2.w * inv);
            dst[12] = (_Float16)(f3.x * inv); dst[13] = (_Float16)(f3.y * inv);
            dst[14] = (_Float16)(f3.z * inv); dst[15] = (_Float16)(f3.w * inv);
        }
    }

    int cur = 0;
    for (int kc = 0; kc < DD / 32; ++kc) {
        __syncthreads();   // Bt[cur] visible; Bt[1-cur] free to overwrite

        if (kc + 1 < DD / 32) {   // prefetch next K-chunk into the other buffer
#pragma unroll
            for (int c = 0; c < 2; ++c) {
                int n = c ? ld_n1 : ld_n0;
                if (c && tid >= 128) break;
                int gn  = n0 + n;
                int gnc = gn < NTRAIN ? gn : NTRAIN - 1;
                float inv = gn < NTRAIN ? invn[l * NTRAIN + gnc] : 0.f;
                const float* gp = train + (((size_t)l * NTRAIN + (size_t)gnc) * DD
                                           + (kc + 1) * 32 + ld_kq);
                float4 f0 = ((const float4*)gp)[0], f1 = ((const float4*)gp)[1];
                float4 f2 = ((const float4*)gp)[2], f3 = ((const float4*)gp)[3];
                if (kc + 2 < DD / 32) __builtin_prefetch(gp + 32, 0, 1);
                _Float16* dst = &Bt[1 - cur][n * BT_STRIDE + ld_kq];
                dst[0]  = (_Float16)(f0.x * inv); dst[1]  = (_Float16)(f0.y * inv);
                dst[2]  = (_Float16)(f0.z * inv); dst[3]  = (_Float16)(f0.w * inv);
                dst[4]  = (_Float16)(f1.x * inv); dst[5]  = (_Float16)(f1.y * inv);
                dst[6]  = (_Float16)(f1.z * inv); dst[7]  = (_Float16)(f1.w * inv);
                dst[8]  = (_Float16)(f2.x * inv); dst[9]  = (_Float16)(f2.y * inv);
                dst[10] = (_Float16)(f2.z * inv); dst[11] = (_Float16)(f2.w * inv);
                dst[12] = (_Float16)(f3.x * inv); dst[13] = (_Float16)(f3.y * inv);
                dst[14] = (_Float16)(f3.z * inv); dst[15] = (_Float16)(f3.w * inv);
            }
        }

        // A fragment (lane half j -> K = hi*16 + j, contiguous, matching B)
        const _Float16* ap = &Aq[(mt * 16 + ln) * AQ_STRIDE + kc * 32 + hi * 16];
        v8h a0 = *(const v8h*)ap;
        v8h a1 = *(const v8h*)(ap + 8);
        v16h a = __builtin_shufflevector(a0, a1, 0,1,2,3,4,5,6,7,8,9,10,11,12,13,14,15);

        // B fragments (this code's order: lane half j -> K = hi*16 + j, contiguous, matching A)
        const _Float16* bb = &Bt[cur][0];
        const _Float16* bp0 = bb + ((ntb + 0) * 16 + ln) * BT_STRIDE + hi * 16;
        const _Float16* bp1 = bb + ((ntb + 1) * 16 + ln) * BT_STRIDE + hi * 16;
        const _Float16* bp2 = bb + ((ntb + 2) * 16 + ln) * BT_STRIDE + hi * 16;
        v8h t0 = *(const v8h*)bp0, t1 = *(const v8h*)(bp0 + 8);
        v16h b0 = __builtin_shufflevector(t0, t1, 0,1,2,3,4,5,6,7,8,9,10,11,12,13,14,15);
        v8h t2 = *(const v8h*)bp1, t3 = *(const v8h*)(bp1 + 8);
        v16h b1 = __builtin_shufflevector(t2, t3, 0,1,2,3,4,5,6,7,8,9,10,11,12,13,14,15);
        v8h t4 = *(const v8h*)bp2, t5 = *(const v8h*)(bp2 + 8);
        v16h b2 = __builtin_shufflevector(t4, t5, 0,1,2,3,4,5,6,7,8,9,10,11,12,13,14,15);

        acc0 = __builtin_amdgcn_wmma_f32_16x16x32_f16(false, a, false, b0, (short)0, acc0, false, false);
        acc1 = __builtin_amdgcn_wmma_f32_16x16x32_f16(false, a, false, b1, (short)0, acc1, false, false);
        acc2 = __builtin_amdgcn_wmma_f32_16x16x32_f16(false, a, false, b2, (short)0, acc2, false, false);

        cur ^= 1;
    }

    // epilogue: C/D layout N=lane&15, M = r + 8*hi
#pragma unroll
    for (int r = 0; r < 8; ++r) {
        int m = m0 + mt * 16 + hi * 8 + r;
        size_t o = ((size_t)(l * BQ + m)) * NPAD + (size_t)n0;
        keys[o + (ntb + 0) * 16 + ln] = sortkey16(-acc0[r]);
        keys[o + (ntb + 1) * 16 + ln] = sortkey16(-acc1[r]);
        keys[o + (ntb + 2) * 16 + ln] = sortkey16(-acc2[r]);
    }
}

// ---- pass 3: per (l,b) radix-select top-K and count neighbor labels ----
__global__ __launch_bounds__(256) void select_count(const unsigned short* __restrict__ keys,
                                                    const int* __restrict__ labels,
                                                    int* __restrict__ counts) {
    __shared__ unsigned int hist[256];
    __shared__ int lc[NC];
    __shared__ int sCoarse, sBelow0, sT, sNeed;
    __shared__ unsigned int tieCnt;
    __shared__ int tieN[TIE_CAP];

    const int bid = blockIdx.x;          // 0..511
    const int l = bid >> 8, b = bid & 255;
    const unsigned short* row = keys + ((size_t)(l * BQ + b)) * NPAD;
    const int tid = threadIdx.x;

    hist[tid] = 0;
    __syncthreads();
    for (int i = tid; i < NTRAIN; i += 256) atomicAdd(&hist[row[i] >> 8], 1u);
    __syncthreads();
    if (tid == 0) {
        unsigned run = 0; int cb = 255;
        for (int c = 0; c < 256; ++c) {
            if (run + hist[c] >= KNN) { cb = c; break; }
            run += hist[c];
        }
        sCoarse = cb; sBelow0 = (int)run;
    }
    __syncthreads();
    const int coarse = sCoarse;
    hist[tid] = 0;
    __syncthreads();
    for (int i = tid; i < NTRAIN; i += 256) {
        unsigned k = row[i];
        if ((int)(k >> 8) == coarse) atomicAdd(&hist[k & 255u], 1u);
    }
    __syncthreads();
    if (tid == 0) {
        int run = sBelow0; int fb = 255;
        for (int c = 0; c < 256; ++c) {
            if (run + (int)hist[c] >= KNN) { fb = c; break; }
            run += (int)hist[c];
        }
        sT = (coarse << 8) | fb; sNeed = KNN - run;
        tieCnt = 0;
    }
    if (tid < NC) lc[tid] = 0;
    __syncthreads();
    const int T = sT, need = sNeed;
    for (int i = tid; i < NTRAIN; i += 256) {
        int k = (int)row[i];
        if (k < T) {
            atomicAdd(&lc[labels[i]], 1);
        } else if (k == T) {
            unsigned p = atomicAdd(&tieCnt, 1u);
            if (p < TIE_CAP) tieN[p] = i;
        }
    }
    __syncthreads();
    int tc = (int)tieCnt;
    if (tc <= TIE_CAP) {
        // deterministic: among tied keys take the `need` smallest train indices
        for (int idx = tid; idx < tc; idx += 256) {
            int ni = tieN[idx];
            int rank = 0;
            for (int j = 0; j < tc; ++j) rank += (tieN[j] < ni) ? 1 : 0;
            if (rank < need) atomicAdd(&lc[labels[ni]], 1);
        }
    } else {
        for (int idx = tid; idx < need && idx < TIE_CAP; idx += 256)
            atomicAdd(&lc[labels[tieN[idx]]], 1);
    }
    __syncthreads();
    if (tid < NC) atomicAdd(&counts[b * NC + tid], lc[tid]);
}

// ---- pass 4: conformal p-values + credibility ----
__global__ __launch_bounds__(256) void finalize(const int* __restrict__ counts,
                                                const int* __restrict__ cali,
                                                float* __restrict__ out) {
    __shared__ int cs[NBCALI];
    const int tid = threadIdx.x;
    for (int i = tid; i < NBCALI; i += 256) cs[i] = cali[i];
    __syncthreads();
    const int b = tid;                   // exactly 256 threads
    float best = -1.f; int bc = 0;
#pragma unroll
    for (int c = 0; c < NC; ++c) {
        int val = LL * KNN - counts[b * NC + c];
        int lo = 0, hiN = NBCALI;        // searchsorted left
        while (lo < hiN) {
            int mid = (lo + hiN) >> 1;
            if (cs[mid] < val) lo = mid + 1; else hiN = mid;
        }
        float p = (float)(NBCALI - lo) / (float)NBCALI;
        if (p > best) { best = p; bc = c; }
    }
#pragma unroll
    for (int c = 0; c < NC; ++c) out[b * NC + c] = (c == bc) ? best : 0.f;
}

extern "C" void kernel_launch(void* const* d_in, const int* in_sizes, int n_in,
                              void* d_out, int out_size, void* d_ws, size_t ws_size,
                              hipStream_t stream) {
    (void)in_sizes; (void)n_in; (void)out_size; (void)ws_size;
    const float* train  = (const float*)d_in[0];   // [L,N,D] f32
    const float* query  = (const float*)d_in[1];   // [L,B,D] f32
    const int*   labels = (const int*)d_in[2];     // [N]
    const int*   cali   = (const int*)d_in[3];     // [750] sorted
    float*       out    = (float*)d_out;           // [B,C] f32

    char* ws = (char*)d_ws;
    float*          invn   = (float*)(ws + 0);                    // 200000*4 = 800000 B
    _Float16*       qh     = (_Float16*)(ws + 800256);            // 512*512*2 = 524288 B
    int*            counts = (int*)(ws + 800256 + 524288);        // 2560*4    = 10240 B
    unsigned short* keys   = (unsigned short*)(ws + 1334784);     // 512*100032*2 B

    prep_train <<<dim3((LL * NTRAIN) / 8), dim3(256), 0, stream>>>(train, invn);
    prep_query <<<dim3((LL * BQ) / 8),     dim3(256), 0, stream>>>(query, qh);
    init_counts<<<dim3((BQ * NC + 255) / 256), dim3(256), 0, stream>>>(counts);
    gemm_keys  <<<dim3(NPAD / NBLK, BQ / MBLK, LL), dim3(256), 0, stream>>>(train, qh, invn, keys);
    select_count<<<dim3(LL * BQ), dim3(256), 0, stream>>>(keys, labels, counts);
    finalize   <<<dim3(1), dim3(256), 0, stream>>>(counts, cali, out);
}
